// GraphLSTMClassifier_30657476559142
// MI455X (gfx1250) — compile-verified, matrix operand order checked by reading, not run on hardware
//
#include <hip/hip_runtime.h>
#include <hip/hip_bf16.h>

typedef __attribute__((ext_vector_type(16))) _Float16 v16h;
typedef __attribute__((ext_vector_type(8)))  _Float16 v8h;
typedef __attribute__((ext_vector_type(8)))  float    v8f;

#define T_MAX  200
#define NEG_SLOPE 0.2f

// ---------- helpers ----------
__device__ __forceinline__ unsigned f2ord(float f) {
    unsigned u = __float_as_uint(f);
    return (u & 0x80000000u) ? ~u : (u | 0x80000000u);
}
__device__ __forceinline__ float ord2f(unsigned u) {
    return (u & 0x80000000u) ? __uint_as_float(u & 0x7FFFFFFFu) : __uint_as_float(~u);
}
__device__ __forceinline__ float sigf(float x) { return 1.0f / (1.0f + __expf(-x)); }

// K-index inside a 16-bit A/B fragment for element j, given lane
__device__ __forceinline__ int kmap(int lane, int j) {
    int kbase = ((lane >> 4) & 1) * 8;
    return kbase + (j < 8 ? j : j + 8);
}
__device__ __forceinline__ v16h cat8(v8h lo, v8h hi) {
    return __builtin_shufflevector(lo, hi, 0, 1, 2, 3, 4, 5, 6, 7,
                                   8, 9, 10, 11, 12, 13, 14, 15);
}

// ---------- generic fill ----------
__global__ void fill_u32(unsigned* __restrict__ p, unsigned v, long n) {
    long g = (long)blockIdx.x * blockDim.x + threadIdx.x;
    if (g < n) p[g] = v;
}

// ---------- f32 -> f16 convert ----------
__global__ void cvt_f16(const float* __restrict__ in, _Float16* __restrict__ o, long n) {
    long g = (long)blockIdx.x * blockDim.x + threadIdx.x;
    if (g < n) o[g] = (_Float16)in[g];
}

// ---------- pack a 128x128 W into WMMA B-fragment order (kb 0..3, nt 0..7) ----------
__global__ void pack_w128(const float* __restrict__ W, _Float16* __restrict__ wp) {
    int g = blockIdx.x * blockDim.x + threadIdx.x;   // 0..16383
    if (g >= 32 * 512) return;
    int j    = g & 15;
    int lane = (g >> 4) & 31;
    int frag = g >> 9;            // kb*8 + nt
    int kb   = frag >> 3;
    int nt   = frag & 7;
    int col  = nt * 16 + (lane & 15);
    int k    = kb * 32 + kmap(lane, j);
    wp[g] = (_Float16)W[k * 128 + col];
}

// ---------- node GEMM: out[M x 128] = in[M x 128] @ W[128 x 128] (packed f16 B) ----------
__global__ void __launch_bounds__(256) gemm_nodes(const float* __restrict__ in,
                                                  const _Float16* __restrict__ wp,
                                                  float* __restrict__ out, int nrows) {
    int lane  = threadIdx.x & 31;
    int nt    = threadIdx.x >> 5;            // 0..7 (wave = N tile)
    int m0    = blockIdx.x * 16;
    int l16   = lane & 15;
    int lhi   = (lane >> 4) & 1;
    int kbase = lhi * 8;
    int row   = m0 + l16;
    int arow  = row < nrows ? row : 0;       // clamp: garbage rows never stored
    const float* ap = in + (long)arow * 128;
    int col = nt * 16 + l16;
    v8f acc = {};
#pragma unroll
    for (int kb = 0; kb < 4; ++kb) {
        int k0 = kb * 32 + kbase;
        float4 q0 = *(const float4*)(ap + k0);
        float4 q1 = *(const float4*)(ap + k0 + 4);
        float4 q2 = *(const float4*)(ap + k0 + 16);
        float4 q3 = *(const float4*)(ap + k0 + 20);
        v16h a;
        a[0]  = (_Float16)q0.x; a[1]  = (_Float16)q0.y;
        a[2]  = (_Float16)q0.z; a[3]  = (_Float16)q0.w;
        a[4]  = (_Float16)q1.x; a[5]  = (_Float16)q1.y;
        a[6]  = (_Float16)q1.z; a[7]  = (_Float16)q1.w;
        a[8]  = (_Float16)q2.x; a[9]  = (_Float16)q2.y;
        a[10] = (_Float16)q2.z; a[11] = (_Float16)q2.w;
        a[12] = (_Float16)q3.x; a[13] = (_Float16)q3.y;
        a[14] = (_Float16)q3.z; a[15] = (_Float16)q3.w;
        v16h b = *(const v16h*)(wp + (kb * 8 + nt) * 512 + lane * 16);
        acc = __builtin_amdgcn_wmma_f32_16x16x32_f16(false, a, false, b, (short)0,
                                                     acc, false, false);
    }
#pragma unroll
    for (int r = 0; r < 8; ++r) {
        int orow = m0 + r + lhi * 8;
        if (orow < nrows) out[(long)orow * 128 + col] = acc[r];
    }
}

// ---------- per-node attention logits ----------
__global__ void node_alpha(const float* __restrict__ h, const float* __restrict__ asv,
                           const float* __restrict__ adv, float* __restrict__ as_,
                           float* __restrict__ ad_, int n) {
    int wid  = (int)(((long)blockIdx.x * blockDim.x + threadIdx.x) >> 5);
    int lane = threadIdx.x & 31;
    if (wid >= n) return;
    float s = 0.f, d = 0.f;
#pragma unroll
    for (int i = 0; i < 4; ++i) {
        float v = h[(long)wid * 128 + lane + 32 * i];
        s += v * asv[lane + 32 * i];
        d += v * adv[lane + 32 * i];
    }
#pragma unroll
    for (int off = 16; off; off >>= 1) { s += __shfl_xor(s, off); d += __shfl_xor(d, off); }
    if (lane == 0) { as_[wid] = s; ad_[wid] = d; }
}

// ---------- GAT edge logits + segment max (edges E..E+n-1 are self loops) ----------
__global__ void edge_logits(const int* __restrict__ src, const int* __restrict__ dst,
                            int E, int n, const float* __restrict__ as_,
                            const float* __restrict__ ad_, float* __restrict__ e,
                            unsigned* __restrict__ mo) {
    long g = (long)blockIdx.x * blockDim.x + threadIdx.x;
    if (g >= (long)E + n) return;
    int s = g < E ? src[g] : (int)(g - E);
    int d = g < E ? dst[g] : (int)(g - E);
    float v = as_[s] + ad_[d];
    v = v > 0.f ? v : NEG_SLOPE * v;
    e[g] = v;
    atomicMax(&mo[d], f2ord(v));
}

// ---------- exp(e - max) + segment sum ----------
__global__ void edge_exp(const int* __restrict__ dst, int E, int n,
                         float* __restrict__ e, const unsigned* __restrict__ mo,
                         float* __restrict__ denom) {
    long g = (long)blockIdx.x * blockDim.x + threadIdx.x;
    if (g >= (long)E + n) return;
    int d = g < E ? dst[g] : (int)(g - E);
    float x = expf(e[g] - ord2f(mo[d]));
    e[g] = x;
    atomicAdd(&denom[d], x);
}

// ---------- GAT weighted scatter ----------
__global__ void gat_scatter(const int* __restrict__ src, const int* __restrict__ dst,
                            int E, int n, const float* __restrict__ ex,
                            const float* __restrict__ denom, const float* __restrict__ h,
                            float* __restrict__ out) {
    long g = (long)blockIdx.x * blockDim.x + threadIdx.x;
    long edge = g >> 5; int lane = (int)(g & 31);
    if (edge >= (long)E + n) return;
    int s = edge < E ? src[edge] : (int)(edge - E);
    int d = edge < E ? dst[edge] : (int)(edge - E);
    float alpha = ex[edge] / denom[d];
    int f0 = lane * 4;
    const float4 hv = *(const float4*)&h[(long)s * 128 + f0];
    atomicAdd(&out[(long)d * 128 + f0 + 0], hv.x * alpha);
    atomicAdd(&out[(long)d * 128 + f0 + 1], hv.y * alpha);
    atomicAdd(&out[(long)d * 128 + f0 + 2], hv.z * alpha);
    atomicAdd(&out[(long)d * 128 + f0 + 3], hv.w * alpha);
}

// ---------- bias + relu in place ----------
__global__ void bias_relu(float* __restrict__ x, const float* __restrict__ b, long n) {
    long g = (long)blockIdx.x * blockDim.x + threadIdx.x;
    if (g >= n) return;
    float v = x[g] + b[g & 127];
    x[g] = v > 0.f ? v : 0.f;
}

// ---------- degree (incl self loops) ----------
__global__ void deg_kernel(const int* __restrict__ dst, int E, int n, float* __restrict__ deg) {
    long g = (long)blockIdx.x * blockDim.x + threadIdx.x;
    if (g >= (long)E + n) return;
    int d = g < E ? dst[g] : (int)(g - E);
    atomicAdd(&deg[d], 1.0f);
}
__global__ void rsqrt_kernel(float* __restrict__ p, int n) {
    int g = blockIdx.x * blockDim.x + threadIdx.x;
    if (g < n) p[g] = rsqrtf(fmaxf(p[g], 1.0f));
}

// ---------- GCN scatter ----------
__global__ void gcn_scatter(const int* __restrict__ src, const int* __restrict__ dst,
                            int E, int n, const float* __restrict__ dinv,
                            const float* __restrict__ h, float* __restrict__ out) {
    long g = (long)blockIdx.x * blockDim.x + threadIdx.x;
    long edge = g >> 5; int lane = (int)(g & 31);
    if (edge >= (long)E + n) return;
    int s = edge < E ? src[edge] : (int)(edge - E);
    int d = edge < E ? dst[edge] : (int)(edge - E);
    float coef = dinv[s] * dinv[d];
    int f0 = lane * 4;
    const float4 hv = *(const float4*)&h[(long)s * 128 + f0];
    atomicAdd(&out[(long)d * 128 + f0 + 0], hv.x * coef);
    atomicAdd(&out[(long)d * 128 + f0 + 1], hv.y * coef);
    atomicAdd(&out[(long)d * 128 + f0 + 2], hv.z * coef);
    atomicAdd(&out[(long)d * 128 + f0 + 3], hv.w * coef);
}

// ---------- graph bookkeeping ----------
__global__ void count_kernel(const int* __restrict__ batch, int n, int* __restrict__ counts) {
    int g = blockIdx.x * blockDim.x + threadIdx.x;
    if (g < n) atomicAdd(&counts[batch[g]], 1);
}
__global__ void scan_kernel(const int* __restrict__ counts, int* __restrict__ starts, int nG) {
    if (threadIdx.x == 0 && blockIdx.x == 0) {
        int acc = 0;
        for (int i = 0; i < nG; ++i) { starts[i] = acc; acc += counts[i]; }
    }
}

// ---------- pack [W_ih;W_hh]^T into WMMA B-fragment order (kb 0..7, nt 0..31) ----------
__global__ void pack_w(const float* __restrict__ W_ih, const float* __restrict__ W_hh,
                       _Float16* __restrict__ wp) {
    int g = blockIdx.x * blockDim.x + threadIdx.x;
    if (g >= 256 * 512) return;
    int j    = g & 15;
    int lane = (g >> 4) & 31;
    int frag = g >> 9;
    int kb   = frag >> 5;
    int nt   = frag & 31;
    int col  = nt * 16 + (lane & 15);          // gate-row n in [0,512)
    int k    = kb * 32 + kmap(lane, j);        // concat K in [0,256)
    float v  = (k < 128) ? W_ih[col * 128 + k] : W_hh[col * 128 + (k - 128)];
    wp[g] = (_Float16)v;
}

// ---------- persistent single-WGP LSTM ----------
__global__ void __launch_bounds__(1024) lstm_kernel(
        const _Float16* __restrict__ xh, const int* __restrict__ starts,
        const int* __restrict__ lens, const _Float16* __restrict__ wp,
        const float* __restrict__ b_ih, const float* __restrict__ b_hh,
        float* __restrict__ grep, int T, int nG, int Ntot) {
    __shared__ _Float16 hbuf[256 * 128];      // 64 KB of the 320 KB WGP LDS
    int tid  = threadIdx.x;
    int lane = tid & 31;
    int wave = tid >> 5;
    int mt   = wave & 15;
    int hg   = wave >> 4;
    int l16  = lane & 15;
    int lhi  = (lane >> 4) & 1;
    int kbase = lhi * 8;
    int row  = mt * 16 + l16;                  // A-matrix row (graph index)

    for (int i = tid; i < 256 * 128; i += 1024) hbuf[i] = (_Float16)0.0f;
    __syncthreads();

    float c[4][8];
#pragma unroll
    for (int j = 0; j < 4; ++j)
#pragma unroll
        for (int r = 0; r < 8; ++r) c[j][r] = 0.0f;

    int rstart = (row < nG) ? starts[row] : 0;

    int bB[8], lenB[8];
#pragma unroll
    for (int r = 0; r < 8; ++r) {
        bB[r]   = mt * 16 + lhi * 8 + r;       // C/D row mapping
        lenB[r] = (bB[r] < nG) ? lens[bB[r]] : -1;
    }

    // hoisted biases: bsum[j][gate] for ncol = (hg*4+j)*16 + l16
    float bsum[4][4];
#pragma unroll
    for (int j = 0; j < 4; ++j) {
        int ncol = (hg * 4 + j) * 16 + l16;
#pragma unroll
        for (int gidx = 0; gidx < 4; ++gidx)
            bsum[j][gidx] = b_ih[gidx * 128 + ncol] + b_hh[gidx * 128 + ncol];
    }

    const _Float16* hrow = &hbuf[row * 128];

    for (int t = 0; t < T; ++t) {
        // clamp node index: out-of-range / finished rows read harmless garbage that
        // only affects their own (never-read-again) batch row
        int node = rstart + t;
        if (node >= Ntot) node = 0;
        const _Float16* xrow = xh + (long)node * 128;

        // --- load all A fragments (x features + old h from LDS) ---
        v16h afr[8];
#pragma unroll
        for (int kb = 0; kb < 4; ++kb) {
            int k0 = kb * 32 + kbase;
            v8h lo = *(const v8h*)(xrow + k0);
            v8h hi = *(const v8h*)(xrow + k0 + 16);
            afr[kb] = cat8(lo, hi);
        }
#pragma unroll
        for (int kb = 4; kb < 8; ++kb) {
            int k0 = (kb - 4) * 32 + kbase;
            v8h lo = *(const v8h*)(hrow + k0);
            v8h hi = *(const v8h*)(hrow + k0 + 16);
            afr[kb] = cat8(lo, hi);
        }
        __syncthreads();   // all reads of old h complete

#pragma unroll 1
        for (int j = 0; j < 4; ++j) {
            int ht = hg * 4 + j;
            v8f ai = {}, af = {}, ag = {}, ao = {};
#pragma unroll
            for (int kb = 0; kb < 8; ++kb) {
                const _Float16* fb = wp + (long)kb * 32 * 512 + lane * 16;
                v16h bi = *(const v16h*)(fb + (long)(0 * 8 + ht) * 512);
                v16h bf = *(const v16h*)(fb + (long)(1 * 8 + ht) * 512);
                v16h bg = *(const v16h*)(fb + (long)(2 * 8 + ht) * 512);
                v16h bo = *(const v16h*)(fb + (long)(3 * 8 + ht) * 512);
                ai = __builtin_amdgcn_wmma_f32_16x16x32_f16(false, afr[kb], false, bi,
                                                            (short)0, ai, false, false);
                af = __builtin_amdgcn_wmma_f32_16x16x32_f16(false, afr[kb], false, bf,
                                                            (short)0, af, false, false);
                ag = __builtin_amdgcn_wmma_f32_16x16x32_f16(false, afr[kb], false, bg,
                                                            (short)0, ag, false, false);
                ao = __builtin_amdgcn_wmma_f32_16x16x32_f16(false, afr[kb], false, bo,
                                                            (short)0, ao, false, false);
            }
            // --- pointwise LSTM cell update for this (mt, ht) tile ---
            int ncol = ht * 16 + l16;
#pragma unroll
            for (int r = 0; r < 8; ++r) {
                float gi = ai[r] + bsum[j][0];
                float gf = af[r] + bsum[j][1];
                float gg = ag[r] + bsum[j][2];
                float go = ao[r] + bsum[j][3];
                float cc = sigf(gf) * c[j][r] + sigf(gi) * tanhf(gg);
                c[j][r] = cc;
                float hv = sigf(go) * tanhf(cc);
                int b = bB[r];
                hbuf[b * 128 + ncol] = (_Float16)hv;
                if (t == lenB[r] - 1) grep[b * 128 + ncol] = hv;
            }
        }
        __syncthreads();   // new h visible before next step's reads
    }
}

// ---------- final FC ----------
__global__ void fc_kernel(const float* __restrict__ grep, const float* __restrict__ W,
                          const float* __restrict__ b, float* __restrict__ out, int nG) {
    int g = blockIdx.x * blockDim.x + threadIdx.x;
    if (g >= nG * 2) return;
    int gb = g >> 1, o = g & 1;
    float s = b[o];
    for (int h = 0; h < 128; ++h) s += grep[gb * 128 + h] * W[h * 2 + o];
    out[g] = s;
}

static inline int cdiv(long a, long b) { return (int)((a + b - 1) / b); }

extern "C" void kernel_launch(void* const* d_in, const int* in_sizes, int n_in,
                              void* d_out, int out_size, void* d_ws, size_t ws_size,
                              hipStream_t stream) {
    const float* x     = (const float*)d_in[0];
    const int*   ei    = (const int*)d_in[1];
    const int*   batch = (const int*)d_in[2];
    const float* W_gat = (const float*)d_in[3];
    const float* a_src = (const float*)d_in[4];
    const float* a_dst = (const float*)d_in[5];
    const float* b_gat = (const float*)d_in[6];
    const float* W_gcn = (const float*)d_in[7];
    const float* b_gcn = (const float*)d_in[8];
    const float* W_ih  = (const float*)d_in[9];
    const float* W_hh  = (const float*)d_in[10];
    const float* b_ih  = (const float*)d_in[11];
    const float* b_hh  = (const float*)d_in[12];
    const float* W_fc  = (const float*)d_in[13];
    const float* b_fc  = (const float*)d_in[14];
    float* out = (float*)d_out;

    const int N  = in_sizes[0] / 128;
    const int E  = in_sizes[1] / 2;
    const int nG = out_size / 2;         // 250
    const int T  = T_MAX;
    const long EN = (long)E + N;

    const int* esrc = ei;
    const int* edst = ei + E;

    // workspace carving (256-byte aligned regions)
    char* cur = (char*)d_ws;
    auto alloc = [&](size_t bytes) {
        char* p = cur;
        cur += (bytes + 255) & ~(size_t)255;
        return (void*)p;
    };
    float*     F1     = (float*)alloc((size_t)N * 128 * 4);   // h1 / hg
    float*     F2     = (float*)alloc((size_t)N * 128 * 4);   // agg buffers / LSTM input
    float*     as_    = (float*)alloc((size_t)N * 4);
    float*     ad_    = (float*)alloc((size_t)N * 4);
    float*     ebuf   = (float*)alloc((size_t)EN * 4);
    unsigned*  m_ord  = (unsigned*)alloc((size_t)N * 4);
    float*     denom  = (float*)alloc((size_t)N * 4);
    float*     dinv   = (float*)alloc((size_t)N * 4);
    int*       counts = (int*)alloc(1024);
    int*       starts = (int*)alloc(1024);
    _Float16*  wpack  = (_Float16*)alloc((size_t)256 * 512 * 2);   // LSTM weights
    _Float16*  wg16   = (_Float16*)alloc((size_t)32 * 512 * 2);    // packed W_gat
    _Float16*  wc16   = (_Float16*)alloc((size_t)32 * 512 * 2);    // packed W_gcn
    _Float16*  xh     = (_Float16*)alloc((size_t)N * 128 * 2);     // f16 LSTM input
    float*     grep   = (float*)alloc((size_t)nG * 128 * 4);

    // ---- zero init of accumulated state ----
    fill_u32<<<cdiv(N, 256), 256, 0, stream>>>(m_ord, 0u, N);
    fill_u32<<<cdiv(N, 256), 256, 0, stream>>>((unsigned*)denom, 0u, N);
    fill_u32<<<cdiv(N, 256), 256, 0, stream>>>((unsigned*)dinv, 0u, N);
    fill_u32<<<1, 256, 0, stream>>>((unsigned*)counts, 0u, 256);
    fill_u32<<<cdiv((long)nG * 128, 256), 256, 0, stream>>>((unsigned*)grep, 0u, (long)nG * 128);

    // ---- GAT ----
    pack_w128<<<64, 256, 0, stream>>>(W_gat, wg16);
    gemm_nodes<<<cdiv(N, 16), 256, 0, stream>>>(x, wg16, F1, N);
    node_alpha<<<cdiv(N, 8), 256, 0, stream>>>(F1, a_src, a_dst, as_, ad_, N);
    edge_logits<<<cdiv(EN, 256), 256, 0, stream>>>(esrc, edst, E, N, as_, ad_, ebuf, m_ord);
    edge_exp<<<cdiv(EN, 256), 256, 0, stream>>>(edst, E, N, ebuf, m_ord, denom);
    fill_u32<<<cdiv((long)N * 128, 256), 256, 0, stream>>>((unsigned*)F2, 0u, (long)N * 128);
    gat_scatter<<<cdiv(EN * 32, 256), 256, 0, stream>>>(esrc, edst, E, N, ebuf, denom, F1, F2);
    bias_relu<<<cdiv((long)N * 128, 256), 256, 0, stream>>>(F2, b_gat, (long)N * 128);

    // ---- GCN ----
    pack_w128<<<64, 256, 0, stream>>>(W_gcn, wc16);
    gemm_nodes<<<cdiv(N, 16), 256, 0, stream>>>(F2, wc16, F1, N);
    deg_kernel<<<cdiv(EN, 256), 256, 0, stream>>>(edst, E, N, dinv);
    rsqrt_kernel<<<cdiv(N, 256), 256, 0, stream>>>(dinv, N);
    fill_u32<<<cdiv((long)N * 128, 256), 256, 0, stream>>>((unsigned*)F2, 0u, (long)N * 128);
    gcn_scatter<<<cdiv(EN * 32, 256), 256, 0, stream>>>(esrc, edst, E, N, dinv, F1, F2);
    bias_relu<<<cdiv((long)N * 128, 256), 256, 0, stream>>>(F2, b_gcn, (long)N * 128);

    // ---- dense-batch bookkeeping ----
    count_kernel<<<cdiv(N, 256), 256, 0, stream>>>(batch, N, counts);
    scan_kernel<<<1, 32, 0, stream>>>(counts, starts, nG);

    // ---- LSTM ----
    pack_w<<<cdiv(256 * 512, 256), 256, 0, stream>>>(W_ih, W_hh, wpack);
    cvt_f16<<<cdiv((long)N * 128, 256), 256, 0, stream>>>(F2, xh, (long)N * 128);
    lstm_kernel<<<1, 1024, 0, stream>>>(xh, starts, counts, wpack, b_ih, b_hh,
                                        grep, T, nG, N);

    // ---- FC ----
    fc_kernel<<<cdiv(nG * 2, 256), 256, 0, stream>>>(grep, W_fc, b_fc, out, nG);
}
